// ObjectRelation_1_67662914781662
// MI455X (gfx1250) — matrix operand order repaired and checked by reading, not compiled, not run on hardware
//
#include <hip/hip_runtime.h>
#include <hip/hip_bf16.h>

// ---------------------------------------------------------------------------
// Types for CDNA5 WMMA (gfx1250, wave32)
// ---------------------------------------------------------------------------
typedef __attribute__((ext_vector_type(16))) __bf16        v16bf;
typedef __attribute__((ext_vector_type(8)))  float         v8f;
typedef __attribute__((ext_vector_type(8)))  unsigned short u16x8;

union FragCast {
    v16bf v;
    u16x8 h[2];
};

__device__ __forceinline__ unsigned short f2bf(float f) {
    __bf16 b = (__bf16)f;                 // fptrunc, RNE
    return __builtin_bit_cast(unsigned short, b);
}

// LDS tile row stride: 56 ushorts = 112 bytes = 28 banks.
// 16 consecutive rows -> 16 disjoint 4-bank groups for ds_read_b128.
#define LDSTR   56
#define TILE_US (128 * LDSTR)     // ushorts per tile buffer
#define TILE_B  (TILE_US * 2)     // 14336 bytes

// A-fragment (ISA 16-bit A layout): lane half h holds K = {8h..8h+7, 16+8h..+7}
__device__ __forceinline__ v16bf load_frag_a(const unsigned short* S, int row, int half) {
    FragCast u;
    u.h[0] = *reinterpret_cast<const u16x8*>(S + row * LDSTR + half * 8);
    u.h[1] = *reinterpret_cast<const u16x8*>(S + row * LDSTR + 16 + half * 8);
    return u.v;
}

// B-fragment (ISA B layout): lane half h holds K = {8h..8h+7, 16+8h..+7}
__device__ __forceinline__ v16bf load_frag_b(const unsigned short* S, int row, int half) {
    FragCast u;
    u.h[0] = *reinterpret_cast<const u16x8*>(S + row * LDSTR + half * 8);
    u.h[1] = *reinterpret_cast<const u16x8*>(S + row * LDSTR + 16 + half * 8);
    return u.v;
}

// ---------------------------------------------------------------------------
// CDNA5 async global->LDS copy (ASYNCcnt-tracked DMA, bypasses VGPRs).
// INST_OFFSET is added to BOTH the LDS and the global address (ISA 15.18.3),
// so one base pair covers a contiguous 32-byte chunk.
// ---------------------------------------------------------------------------
__device__ __forceinline__ void async_cp_32B(unsigned lds, const void* g) {
    asm volatile(
        "global_load_async_to_lds_b128 %0, %1, off\n\t"
        "global_load_async_to_lds_b128 %0, %1, off offset:16"
        :: "v"(lds), "v"((unsigned long long)(uintptr_t)g)
        : "memory");
}

__device__ __forceinline__ void wait_async0() {
    asm volatile("s_wait_asynccnt 0" ::: "memory");
}

// ---------------------------------------------------------------------------
// bf16-WMMA GEMM:  D[M,Nt] = A[M,K] @ B[Nt,K]^T  (+ bias) (+ resid)
//   A, B: bf16 row-major.  D: f32 (OUTBF=0) or bf16 (OUTBF=1).
//   OUTTR=1: store D transposed as DT[Nt, M].
// Block = 256 threads (8 waves), tile 128x128, K stepped by 32.
// Double-buffered LDS; tile k+1 DMA'd via async-to-LDS during tile-k WMMA.
// M, Nt multiples of 128; K multiple of 32.
// ---------------------------------------------------------------------------
template <bool BIAS, bool RESID, bool OUTBF, bool OUTTR>
__global__ __launch_bounds__(256)
void gemm_bf16_wmma(const unsigned short* __restrict__ A,
                    const unsigned short* __restrict__ B,
                    const float* __restrict__ bias,
                    const float* __restrict__ resid,
                    void* __restrict__ Dv,
                    int M, int Nt, int K) {
    // [buf0: A,B][buf1: A,B]
    __shared__ alignas(16) unsigned short Sm[4 * TILE_US];

    const int tid  = threadIdx.x;
    const int gm   = blockIdx.y * 128;
    const int gn   = blockIdx.x * 128;

    const int w    = tid >> 5;
    const int lane = tid & 31;
    const int half = lane >> 4;
    const int lr   = lane & 15;
    const int wm   = (w >> 1) * 32;   // wave m offset (0,32,64,96)
    const int wn   = (w & 1) * 64;    // wave n offset (0,64)

    v8f acc[2][4];
#pragma unroll
    for (int mi = 0; mi < 2; ++mi)
#pragma unroll
        for (int ni = 0; ni < 4; ++ni)
            acc[mi][ni] = (v8f){0.f, 0.f, 0.f, 0.f, 0.f, 0.f, 0.f, 0.f};

    // staging: 128 rows x 32 bf16; thread moves 32B of A and 32B of B
    const int sr = tid >> 1;           // row 0..127
    const int se = (tid & 1) * 16;     // element offset 0 or 16

    const unsigned short* gA = A + (size_t)(gm + sr) * K + se;
    const unsigned short* gB = B + (size_t)(gn + sr) * K + se;
    const unsigned ldsA = (unsigned)(uintptr_t)(&Sm[0]) + sr * (LDSTR * 2) + se * 2;

    // prologue: DMA tile 0 into buffer 0
    async_cp_32B(ldsA + 0 * TILE_B, gA);
    async_cp_32B(ldsA + 1 * TILE_B, gB);

    const int nk = K >> 5;
    for (int it = 0; it < nk; ++it) {
        const int buf = it & 1;

        wait_async0();        // our tile-it chunks have landed in LDS
        __syncthreads();      // everyone's landed; everyone done reading buf^1

        if (it + 1 < nk) {    // DMA tile it+1 into the other buffer
            const int kk = (it + 1) << 5;
            const unsigned dst = ldsA + (unsigned)(buf ^ 1) * (2 * TILE_B);
            async_cp_32B(dst, gA + kk);
            async_cp_32B(dst + TILE_B, gB + kk);
        }

        const unsigned short* curA = Sm + buf * 2 * TILE_US;
        const unsigned short* curB = curA + TILE_US;

        v16bf af[2], bfr[4];
#pragma unroll
        for (int mi = 0; mi < 2; ++mi)
            af[mi] = load_frag_a(curA, wm + mi * 16 + lr, half);
#pragma unroll
        for (int ni = 0; ni < 4; ++ni)
            bfr[ni] = load_frag_b(curB, wn + ni * 16 + lr, half);

#pragma unroll
        for (int mi = 0; mi < 2; ++mi)
#pragma unroll
            for (int ni = 0; ni < 4; ++ni)
                acc[mi][ni] = __builtin_amdgcn_wmma_f32_16x16x32_bf16(
                    false, af[mi], false, bfr[ni], (short)0, acc[mi][ni],
                    false, false);
    }

    // ---- epilogue (straight-line; all flags compile-time) ----
    float*          Df = (float*)Dv;
    unsigned short* Db = (unsigned short*)Dv;
#pragma unroll
    for (int mi = 0; mi < 2; ++mi) {
#pragma unroll
        for (int ni = 0; ni < 4; ++ni) {
            const int col  = gn + wn + ni * 16 + lr;
            const int row0 = gm + wm + mi * 16 + half * 8;
            float bv = 0.0f;
            if (BIAS) bv = bias[col];
            if (OUTTR) {
                // DT[Nt, M]: lane writes 8 contiguous elements along M
                if (OUTBF) {
                    u16x8 o;
#pragma unroll
                    for (int r = 0; r < 8; ++r) {
                        float v = acc[mi][ni][r];
                        if (BIAS) v += bv;
                        o[r] = f2bf(v);
                    }
                    *reinterpret_cast<u16x8*>(Db + (size_t)col * M + row0) = o;
                } else {
#pragma unroll
                    for (int r = 0; r < 8; ++r) {
                        float v = acc[mi][ni][r];
                        if (BIAS) v += bv;
                        Df[(size_t)col * M + row0 + r] = v;
                    }
                }
            } else {
#pragma unroll
                for (int r = 0; r < 8; ++r) {
                    const size_t idx = (size_t)(row0 + r) * Nt + col;
                    float v = acc[mi][ni][r];
                    if (BIAS)  v += bv;
                    if (RESID) v += resid[idx];
                    if (OUTBF) Db[idx] = f2bf(v);
                    else       Df[idx] = v;
                }
            }
        }
    }
}

// ---------------------------------------------------------------------------
// f32 -> bf16 bulk convert (n multiple of 8)
// ---------------------------------------------------------------------------
__global__ __launch_bounds__(256)
void cvt_f32_bf16(const float* __restrict__ s, unsigned short* __restrict__ d, int n) {
    const int i = (blockIdx.x * 256 + threadIdx.x) * 8;
    if (i >= n) return;
    float4 a = *reinterpret_cast<const float4*>(s + i);
    float4 b = *reinterpret_cast<const float4*>(s + i + 4);
    u16x8 o;
    o[0] = f2bf(a.x); o[1] = f2bf(a.y); o[2] = f2bf(a.z); o[3] = f2bf(a.w);
    o[4] = f2bf(b.x); o[5] = f2bf(b.y); o[6] = f2bf(b.z); o[7] = f2bf(b.w);
    *reinterpret_cast<u16x8*>(d + i) = o;
}

// ---------------------------------------------------------------------------
// Fused: p = softmax(mask_row); t = f_row * p; f_row = softmax(t)
// Writes f32 result (d_out) and bf16 copy (operand of the y GEMM).
// ---------------------------------------------------------------------------
__device__ __forceinline__ float block_reduce_max(float v, float* red, int tid) {
    red[tid] = v;
    __syncthreads();
    for (int s = 128; s > 0; s >>= 1) {
        if (tid < s) red[tid] = fmaxf(red[tid], red[tid + s]);
        __syncthreads();
    }
    float r = red[0];
    __syncthreads();
    return r;
}

__device__ __forceinline__ float block_reduce_sum(float v, float* red, int tid) {
    red[tid] = v;
    __syncthreads();
    for (int s = 128; s > 0; s >>= 1) {
        if (tid < s) red[tid] += red[tid + s];
        __syncthreads();
    }
    float r = red[0];
    __syncthreads();
    return r;
}

__global__ __launch_bounds__(256)
void fuse_mask_softmax(const float* __restrict__ Mlog, float* __restrict__ F,
                       unsigned short* __restrict__ Fb, int N) {
    __shared__ float srow[8192];
    __shared__ float red[256];
    const int tid = threadIdx.x;
    const float*    mrow = Mlog + (size_t)blockIdx.x * N;
    float*          frow = F    + (size_t)blockIdx.x * N;
    unsigned short* brow = Fb   + (size_t)blockIdx.x * N;

    float mx = -3.402823466e38f;
    for (int j = tid; j < N; j += 256) {
        float v = mrow[j];
        srow[j] = v;
        mx = fmaxf(mx, v);
    }
    __syncthreads();
    mx = block_reduce_max(mx, red, tid);

    float s = 0.f;
    for (int j = tid; j < N; j += 256) s += __expf(srow[j] - mx);
    s = block_reduce_sum(s, red, tid);
    const float inv = 1.0f / s;

    float mx2 = -3.402823466e38f;
    for (int j = tid; j < N; j += 256) {
        float t = frow[j] * (__expf(srow[j] - mx) * inv);
        srow[j] = t;
        mx2 = fmaxf(mx2, t);
    }
    __syncthreads();
    mx2 = block_reduce_max(mx2, red, tid);

    float s2 = 0.f;
    for (int j = tid; j < N; j += 256) s2 += __expf(srow[j] - mx2);
    s2 = block_reduce_sum(s2, red, tid);
    const float inv2 = 1.0f / s2;

    for (int j = tid; j < N; j += 256) {
        float v = __expf(srow[j] - mx2) * inv2;
        frow[j] = v;
        brow[j] = f2bf(v);
    }
}

// ---------------------------------------------------------------------------
// Launcher
// ---------------------------------------------------------------------------
extern "C" void kernel_launch(void* const* d_in, const int* in_sizes, int n_in,
                              void* d_out, int out_size, void* d_ws, size_t ws_size,
                              hipStream_t stream) {
    (void)in_sizes; (void)n_in; (void)out_size; (void)ws_size;

    const int N = 8192, C = 2048, IC = 1024;
    const size_t NN  = (size_t)N * N;      // 64M
    const size_t NC  = (size_t)N * C;      // 16M
    const size_t NIC = (size_t)N * IC;     // 8M
    const size_t ICC = (size_t)IC * C;     // 2M

    const float* x       = (const float*)d_in[0];
    const float* theta_w = (const float*)d_in[1];
    const float* theta_b = (const float*)d_in[2];
    const float* phi_w   = (const float*)d_in[3];
    const float* phi_b   = (const float*)d_in[4];
    const float* g_w     = (const float*)d_in[5];
    const float* g_b     = (const float*)d_in[6];
    const float* W_w     = (const float*)d_in[7];
    const float* W_b     = (const float*)d_in[8];
    const float* mask_w  = (const float*)d_in[9];
    const float* mask_b  = (const float*)d_in[10];

    // workspace: [Mlog f32 NN] then bf16 buffers
    char* wsb = (char*)d_ws;
    float* Mlog = (float*)wsb;
    unsigned short* u = (unsigned short*)(wsb + NN * sizeof(float));
    unsigned short* Fbf   = u; u += NN;    // bf16 f_div_C
    unsigned short* xbf   = u; u += NC;
    unsigned short* mwbf  = u; u += NC;
    unsigned short* twbf  = u; u += ICC;
    unsigned short* pwbf  = u; u += ICC;
    unsigned short* gwbf  = u; u += ICC;
    unsigned short* Wwbf  = u; u += ICC;
    unsigned short* thxbf = u; u += NIC;   // theta_x bf16 [N,IC]
    unsigned short* phxbf = u; u += NIC;   // phi_x   bf16 [N,IC]
    unsigned short* gxT   = u; u += NIC;   // g_x^T   bf16 [IC,N]
    unsigned short* ybbf  = u; u += NIC;   // y       bf16 [N,IC]

    float* z   = (float*)d_out;            // [N, C]
    float* Fdc = z + NC;                   // [N, N] f / f_div_C (f32)

    const dim3 blk(256);

    // ---- bulk f32->bf16 converts ----
    cvt_f32_bf16<<<dim3((unsigned)(NC  / 2048)), blk, 0, stream>>>(x,       xbf,  (int)NC);
    cvt_f32_bf16<<<dim3((unsigned)(NC  / 2048)), blk, 0, stream>>>(mask_w,  mwbf, (int)NC);
    cvt_f32_bf16<<<dim3((unsigned)(ICC / 2048)), blk, 0, stream>>>(theta_w, twbf, (int)ICC);
    cvt_f32_bf16<<<dim3((unsigned)(ICC / 2048)), blk, 0, stream>>>(phi_w,   pwbf, (int)ICC);
    cvt_f32_bf16<<<dim3((unsigned)(ICC / 2048)), blk, 0, stream>>>(g_w,     gwbf, (int)ICC);
    cvt_f32_bf16<<<dim3((unsigned)(ICC / 2048)), blk, 0, stream>>>(W_w,     Wwbf, (int)ICC);

    // ---- projections ----
    gemm_bf16_wmma<true, false, true, false><<<dim3(IC / 128, N / 128), blk, 0, stream>>>(
        xbf, twbf, theta_b, nullptr, thxbf, N, IC, C);
    gemm_bf16_wmma<true, false, true, false><<<dim3(IC / 128, N / 128), blk, 0, stream>>>(
        xbf, pwbf, phi_b, nullptr, phxbf, N, IC, C);
    // g projection stored transposed so the y GEMM stays in A@B^T form
    gemm_bf16_wmma<true, false, true, true><<<dim3(IC / 128, N / 128), blk, 0, stream>>>(
        xbf, gwbf, g_b, nullptr, gxT, N, IC, C);

    // ---- mask logits (f32, ws) ----
    gemm_bf16_wmma<true, false, false, false><<<dim3(N / 128, N / 128), blk, 0, stream>>>(
        xbf, mwbf, mask_b, nullptr, Mlog, N, N, C);

    // ---- f = theta_x @ phi_x^T (f32, into d_out's f_div_C region) ----
    gemm_bf16_wmma<false, false, false, false><<<dim3(N / 128, N / 128), blk, 0, stream>>>(
        thxbf, phxbf, nullptr, nullptr, Fdc, N, N, IC);

    // ---- f_div_C = softmax(f * softmax(mask)); also emit bf16 copy ----
    fuse_mask_softmax<<<dim3(N), blk, 0, stream>>>(Mlog, Fdc, Fbf, N);

    // ---- y = f_div_C @ g_x = Fbf @ gxT^T ----
    gemm_bf16_wmma<false, false, true, false><<<dim3(IC / 128, N / 128), blk, 0, stream>>>(
        Fbf, gxT, nullptr, nullptr, ybbf, N, IC, N);

    // ---- z = y @ W_w^T + W_b + x ----
    gemm_bf16_wmma<true, true, false, false><<<dim3(C / 128, N / 128), blk, 0, stream>>>(
        ybbf, Wwbf, W_b, x, z, N, C, IC);
}
